// MozafariMNIST2018_66735201845428
// MI455X (gfx1250) — matrix operand order repaired and checked by reading, not compiled, not run on hardware
//
#include <hip/hip_runtime.h>
#include <hip/hip_bf16.h>

typedef __attribute__((ext_vector_type(16))) _Float16 v16h;
typedef __attribute__((ext_vector_type(8)))  float    v8f;

#define BATCH   4096
#define TSTEPS  15
#define MTOT    (BATCH * TSTEPS)       // 61440 rows
#define K1      784                    // 28*28
#define F1      500
#define F3      10
#define NPAD1   512                    // F1 padded for WMMA tiling
#define KPAD3   512                    // conv3 K padded

#define ASTRIDE 40                     // LDS row stride (halves): 80B rows, 16B aligned,
                                       // 20-bank stride -> conflict-free 16-lane groups
#define BSTRIDE3 520                   // kernel2 w3 LDS stride: 1040B = 65*16

__constant__ const float kConv1T = 26.3424f;  // 15*(784/25)*0.35*0.16

// convert 8 floats (two float4) to f16 and store 16B into LDS
__device__ __forceinline__ void cvt8_store(_Float16* d, float4 p0, float4 p1) {
    d[0] = (_Float16)p0.x; d[1] = (_Float16)p0.y;
    d[2] = (_Float16)p0.z; d[3] = (_Float16)p0.w;
    d[4] = (_Float16)p1.x; d[5] = (_Float16)p1.y;
    d[6] = (_Float16)p1.z; d[7] = (_Float16)p1.w;
}

// gather A (16x32) + two B (32x16) fragments in contiguous k order (k = 16*(lane/16) + s) and issue 2 WMMAs
__device__ __forceinline__ void frag_mma2(const _Float16* __restrict__ As,
                                          const _Float16* __restrict__ Bs,
                                          int lane, int mi, int ni,
                                          v8f& acc0, v8f& acc1) {
    union { v16h v; _Float16 h[16]; } af, bf0, bf1;
    const int mrow = mi * 16 + (lane & 15);
    const int ksh  = (lane >> 4) * 16;         // lanes 16..31: K += 16
    #pragma unroll
    for (int p = 0; p < 8; ++p) {
        const int kk = 2 * p + ksh;  // contiguous: 2x b128
        af.h[2 * p]     = As[mrow * ASTRIDE + kk];
        af.h[2 * p + 1] = As[mrow * ASTRIDE + kk + 1];
    }
    const int nbr = ni * 32 + (lane & 15);
    const int kb  = (lane >> 4) * 16;
    #pragma unroll
    for (int h = 0; h < 16; ++h) {
        bf0.h[h] = Bs[nbr * ASTRIDE + kb + h];
        bf1.h[h] = Bs[(nbr + 16) * ASTRIDE + kb + h];
    }
    acc0 = __builtin_amdgcn_wmma_f32_16x16x32_f16(
               false, af.v, false, bf0.v, (short)0, acc0, false, false);
    acc1 = __builtin_amdgcn_wmma_f32_16x16x32_f16(
               false, af.v, false, bf1.v, (short)0, acc1, false, false);
}

// ---------------------------------------------------------------------------
// Kernel 1: pot1 = input[61440,784] x w1^T[784,500]  (f16 WMMA, f32 accum)
// 256 threads (8 waves); tile 64(M) x 64(N); wave -> 16x32 (2 WMMA, shared A).
// Main K loop (24 iters) is branch-free; guarded tail iteration peeled.
// ---------------------------------------------------------------------------
__global__ __launch_bounds__(256) void conv1_wmma_kernel(
    const float* __restrict__ A,     // input, [MTOT, K1]
    const float* __restrict__ W,     // w1,    [F1,  K1]
    float* __restrict__ spk1,        // [MTOT, F1]
    float* __restrict__ thr1)        // [MTOT, F1]
{
    __shared__ _Float16 As[64 * ASTRIDE];
    __shared__ _Float16 Bs[64 * ASTRIDE];

    const int tid  = threadIdx.x;
    const int lane = tid & 31;
    const int wv   = tid >> 5;       // 0..7
    const int mi   = wv & 3;         // M sub-tile (16 rows)
    const int ni   = wv >> 2;        // N sub-tile (32 cols)
    const int rowBase = blockIdx.x * 64;
    const int colBase = blockIdx.y * 64;

    // staging: 2048 halves per tile / 256 threads = 8 each
    const int lm = tid >> 2;         // A row      0..63
    const int lk = (tid & 3) * 8;    // k offset   0,8,16,24
    // B feature row clamped into range: duplicated rows only feed cols >= 500,
    // which are never stored -> main loop needs no guard at all.
    const int f  = colBase + lm;
    const int fc = (f < F1) ? f : (F1 - 1);

    const float* srcA = A + (rowBase + lm) * (long)K1 + lk;
    const float* srcB = W + fc * (long)K1 + lk;
    _Float16* dA = &As[lm * ASTRIDE + lk];
    _Float16* dB = &Bs[lm * ASTRIDE + lk];

    v8f acc0 = {};
    v8f acc1 = {};

    // ---- main loop: k0 = 0..736, fully in-bounds, no guards ----
    for (int it = 0; it < 24; ++it) {
        float4 a0 = *(const float4*)(srcA);
        float4 a1 = *(const float4*)(srcA + 4);
        float4 b0 = *(const float4*)(srcB);
        float4 b1 = *(const float4*)(srcB + 4);
        __builtin_prefetch(srcA + 32, 0, 3);     // next K chunk of the big stream
        cvt8_store(dA, a0, a1);
        cvt8_store(dB, b0, b1);
        __syncthreads();
        frag_mma2(As, Bs, lane, mi, ni, acc0, acc1);
        __syncthreads();
        srcA += 32;
        srcB += 32;
    }
    // ---- peeled tail: k0 = 768, halves 16..31 are zero padding ----
    {
        float4 a0 = {0.f,0.f,0.f,0.f}, a1 = a0, b0 = a0, b1 = a0;
        if (lk < 16) {                            // k = 768+lk .. 768+lk+7 <= 783
            a0 = *(const float4*)(srcA); a1 = *(const float4*)(srcA + 4);
            b0 = *(const float4*)(srcB); b1 = *(const float4*)(srcB + 4);
        }
        cvt8_store(dA, a0, a1);
        cvt8_store(dB, b0, b1);
        __syncthreads();
        frag_mma2(As, Bs, lane, mi, ni, acc0, acc1);
        __syncthreads();
    }

    // ---- fused fire(): threshold + sign, f32 outputs ----
    #pragma unroll
    for (int j = 0; j < 8; ++j) {
        const int row  = rowBase + mi * 16 + ((lane < 16) ? j : j + 8);
        const int col0 = colBase + ni * 32 + (lane & 15);
        const float pot0 = acc0[j];
        const float pot1 = acc1[j];
        if (col0 < F1) {
            const bool fire = (pot0 >= kConv1T);
            const long o = row * (long)F1 + col0;
            thr1[o] = fire ? pot0 : 0.0f;
            spk1[o] = fire ? 1.0f : 0.0f;
        }
        if (col0 + 16 < F1) {
            const bool fire = (pot1 >= kConv1T);
            const long o = row * (long)F1 + col0 + 16;
            thr1[o] = fire ? pot1 : 0.0f;
            spk1[o] = fire ? 1.0f : 0.0f;
        }
    }
}

// ---------------------------------------------------------------------------
// Kernel 2: pot3 = spk1[61440,500] x w3^T[500,10]  (f16 WMMA; N padded to 16)
// 256 threads (8 waves), 128 rows/block; w3 resident in LDS.
// K loop (16 iters); tail chunk k0 = 480 guarded inside the loop.
// ---------------------------------------------------------------------------
__global__ __launch_bounds__(256) void conv3_wmma_kernel(
    const float* __restrict__ S,     // spk1, [MTOT, F1]
    const float* __restrict__ W3,    // w3,   [F3, F1]
    float* __restrict__ pot3)        // [MTOT, F3]
{
    __shared__ _Float16 Bs[16 * BSTRIDE3];  // w3 as Bs[n][k], n->16, k->512
    __shared__ _Float16 As[128 * ASTRIDE];

    const int tid  = threadIdx.x;
    const int lane = tid & 31;
    const int wv   = tid >> 5;              // 0..7 -> 16-row slice
    const int rowBase = blockIdx.x * 128;

    // preload w3 once (f32 -> f16, zero pad n->16, k->512)
    {
        const int n  = tid >> 4;            // 0..15
        const int kq = (tid & 15) * 32;     // 0..480
        #pragma unroll
        for (int j = 0; j < 32; ++j) {
            const int k = kq + j;
            const float v = (n < F3 && k < F1) ? W3[n * F1 + k] : 0.0f;
            Bs[n * BSTRIDE3 + k] = (_Float16)v;
        }
    }
    __syncthreads();

    v8f acc = {};
    const int lm = tid >> 1;                // 0..127
    const int lk = (tid & 1) * 16;          // 0 or 16
    const float* srcS = S + (rowBase + lm) * (long)F1 + lk;
    _Float16* dA = &As[lm * ASTRIDE + lk];

    const int mrow = wv * 16 + (lane & 15);
    const int ksh  = (lane >> 4) * 16;
    const int nbr  = lane & 15;

    for (int k0 = 0; k0 < KPAD3; k0 += 32) {
        if (k0 + lk + 16 <= F1) {           // constant-folds true for k0 < 480
            float4 p0 = *(const float4*)(srcS);
            float4 p1 = *(const float4*)(srcS + 4);
            float4 p2 = *(const float4*)(srcS + 8);
            float4 p3 = *(const float4*)(srcS + 12);
            cvt8_store(dA,     p0, p1);
            cvt8_store(dA + 8, p2, p3);
        } else {                            // tail: k0=480, lk=16 -> k 496..511
            #pragma unroll
            for (int j = 0; j < 16; ++j) {
                const int k = k0 + lk + j;
                dA[j] = (k < F1) ? (_Float16)srcS[j] : (_Float16)0.0f;
            }
        }
        __syncthreads();

        union { v16h v; _Float16 h[16]; } af, bf;
        #pragma unroll
        for (int p = 0; p < 8; ++p) {
            const int kk = 2 * p + ksh;
            af.h[2 * p]     = As[mrow * ASTRIDE + kk];
            af.h[2 * p + 1] = As[mrow * ASTRIDE + kk + 1];
        }
        const int kb = k0 + (lane >> 4) * 16;
        #pragma unroll
        for (int h = 0; h < 16; ++h) bf.h[h] = Bs[nbr * BSTRIDE3 + kb + h];

        acc = __builtin_amdgcn_wmma_f32_16x16x32_f16(
                  false, af.v, false, bf.v, (short)0, acc, false, false);
        __syncthreads();
        srcS += 32;
    }

    #pragma unroll
    for (int j = 0; j < 8; ++j) {
        const int row = rowBase + wv * 16 + ((lane < 16) ? j : j + 8);
        const int col = lane & 15;
        if (col < F3) pot3[row * (long)F3 + col] = acc[j];
    }
}

// ---------------------------------------------------------------------------
// Kernel 3: winner-take-all. Only t = T-1 matters; winner = argmax_f of
// pot3[b,T-1,f] among strictly positive entries, else -1.
// ---------------------------------------------------------------------------
__global__ __launch_bounds__(256) void winner_kernel(
    const float* __restrict__ pot3,  // [MTOT, F3]
    float* __restrict__ cls)         // [BATCH]
{
    const int b = blockIdx.x * blockDim.x + threadIdx.x;
    if (b >= BATCH) return;
    const float* p = pot3 + ((long)b * TSTEPS + (TSTEPS - 1)) * F3;
    float best = 0.0f;
    int   wi   = -1;
    #pragma unroll
    for (int f = 0; f < F3; ++f) {
        const float v = p[f];
        if (v > 0.0f && v > best) { best = v; wi = f; }
    }
    cls[b] = (float)wi;
}

// ---------------------------------------------------------------------------
extern "C" void kernel_launch(void* const* d_in, const int* in_sizes, int n_in,
                              void* d_out, int out_size, void* d_ws, size_t ws_size,
                              hipStream_t stream)
{
    const float* input = (const float*)d_in[0];   // [B,T,1,28,28]
    const float* w1    = (const float*)d_in[1];   // [500,1,28,28]
    const float* w3    = (const float*)d_in[2];   // [10,500]
    // d_in[3] = max_layer (==3)

    float* out  = (float*)d_out;
    float* spk1 = out;                                    // MTOT*F1
    float* thr1 = out + (long)MTOT * F1;                  // MTOT*F1
    float* pot3 = out + 2L * MTOT * F1;                   // MTOT*F3
    float* cls  = out + 2L * MTOT * F1 + (long)MTOT * F3; // BATCH

    dim3 g1(MTOT / 64, NPAD1 / 64);   // 960 x 8 blocks
    conv1_wmma_kernel<<<g1, 256, 0, stream>>>(input, w1, spk1, thr1);

    conv3_wmma_kernel<<<MTOT / 128, 256, 0, stream>>>(spk1, w3, pot3);

    winner_kernel<<<BATCH / 256, 256, 0, stream>>>(pot3, cls);
}
